// GTransEncoder_12876311954004
// MI455X (gfx1250) — compile-verified, matrix operand order checked by reading, not run on hardware
//
#include <hip/hip_runtime.h>

typedef __bf16 bf16_t;
typedef __attribute__((ext_vector_type(16))) __bf16 v16bf;
typedef __attribute__((ext_vector_type(8)))  __bf16 v8bf;
typedef __attribute__((ext_vector_type(4)))  __bf16 v4bf;
typedef __attribute__((ext_vector_type(8)))  float  v8f;

#define HS     512
#define NMOL   512
#define LMOL   48
#define NA     24577      // 1 + 512*48 atoms
#define NA_PAD 24640      // round up to 64
#define NATT   24576
#define MMSG   53249      // 1 + 512*104 messages
#define MSG_PAD 53312     // round up to 64
#define NHEAD  8
#define DHEAD  64

// ---------------------------------------------------------------------------
// WMMA helpers (CDNA5 wave32, 16x16x32 bf16 -> f32)
// ---------------------------------------------------------------------------
__device__ inline v8f zero8() {
  v8f z;
#pragma unroll
  for (int i = 0; i < 8; ++i) z[i] = 0.f;
  return z;
}

__device__ inline v8f wmma_bf(v16bf a, v16bf b, v8f c) {
  return __builtin_amdgcn_wmma_f32_16x16x32_bf16(false, a, false, b, (short)0, c,
                                                 false, false);
}

// 16x32 fragment load (A row-major MxK, or B from transposed [N][K] storage).
// lane r = lane&15 picks row; lanes 16-31 take the K+8 half; two 16B chunks.
// Requires base 16B-aligned and ld a multiple of 8 elements.
__device__ inline v16bf load_frag(const bf16_t* base, int ld, int lane) {
  int r  = lane & 15;
  int kh = (lane & 16) ? 8 : 0;
  const bf16_t* p = base + (size_t)r * ld;
  v8bf lo = *(const v8bf*)(p + kh);
  v8bf hi = *(const v8bf*)(p + 16 + kh);
  return __builtin_shufflevector(lo, hi, 0, 1, 2, 3, 4, 5, 6, 7,
                                 8, 9, 10, 11, 12, 13, 14, 15);
}

// ---------------------------------------------------------------------------
// Weight convert+transpose: f32 [K x N] row-major -> bf16 [N x Kp], K zero-pad.
// ---------------------------------------------------------------------------
__global__ __launch_bounds__(256) void wconv_kernel(
    const float* __restrict__ in, int ldin, int K, int N,
    bf16_t* __restrict__ outT, int Kp) {
  int idx = blockIdx.x * 256 + threadIdx.x;
  if (idx >= N * Kp) return;
  int n = idx / Kp, k = idx % Kp;
  float v = (k < K) ? in[(size_t)k * ldin + n] : 0.f;
  outT[idx] = (bf16_t)v;
}

// f32 [rows x cols] -> bf16 [rows_pad x cols_pad], zero-padded both dims.
__global__ __launch_bounds__(256) void convpad_kernel(
    const float* __restrict__ in, bf16_t* __restrict__ out,
    int rows, int cols, int colsp, long total) {
  long i = (long)blockIdx.x * 256 + threadIdx.x;
  if (i >= total) return;
  int r = (int)(i / colsp), c = (int)(i % colsp);
  float v = (r < rows && c < cols) ? in[(size_t)r * cols + c] : 0.f;
  out[i] = (bf16_t)v;
}

__global__ __launch_bounds__(256) void zero32_kernel(unsigned int* __restrict__ p,
                                                     long n) {
  long i = (long)blockIdx.x * 256 + threadIdx.x;
  if (i < n) p[i] = 0u;
}

// ---------------------------------------------------------------------------
// Generic WMMA GEMM: C[M,N] = act(A[Mpad,K]_bf16 @ BT[N,Kp]_bf16 (+prev)(+bias))
// A buffers are row-padded to 64 and K-padded to 32 => no bounds checks, no
// LDS, no barriers. Block tile 64x128, 8 waves; each wave owns a 2x2 grid of
// 16x16 sub-tiles (32 rows x 32 cols) so each K-step is 8 global_load_b128
// feeding 4 v_wmma (A and B frags each reused twice in registers).
// ---------------------------------------------------------------------------
__global__ __launch_bounds__(256) void gemm_bf16_kernel(
    const bf16_t* __restrict__ A, int lda, int M, int K,
    const bf16_t* __restrict__ BT, int ldk, int N,
    const float* __restrict__ bias, int act, int accum,
    float* __restrict__ outF, bf16_t* __restrict__ outB) {
  int tile_m = blockIdx.x * 64;
  int tile_n = blockIdx.y * 128;
  int tid = threadIdx.x;
  int wave = tid >> 5, lane = tid & 31;
  int mg = (wave & 1) * 32;   // wave's m-offset within tile (2 sub-tiles)
  int ng = (wave >> 1) * 32;  // wave's n-offset within tile (2 sub-tiles)

  const bf16_t* A0 = A + (size_t)(tile_m + mg) * lda;
  const bf16_t* A1 = A + (size_t)(tile_m + mg + 16) * lda;
  const bf16_t* B0 = BT + (size_t)(tile_n + ng) * ldk;
  const bf16_t* B1 = BT + (size_t)(tile_n + ng + 16) * ldk;

  v8f acc00 = zero8(), acc01 = zero8(), acc10 = zero8(), acc11 = zero8();
  int ksteps = K >> 5;  // K is a multiple of 32 (padded)
  for (int ks = 0; ks < ksteps; ++ks) {
    int k0 = ks << 5;
    v16bf a0 = load_frag(A0 + k0, lda, lane);
    v16bf a1 = load_frag(A1 + k0, lda, lane);
    v16bf b0 = load_frag(B0 + k0, ldk, lane);
    v16bf b1 = load_frag(B1 + k0, ldk, lane);
    acc00 = wmma_bf(a0, b0, acc00);
    acc01 = wmma_bf(a0, b1, acc01);
    acc10 = wmma_bf(a1, b0, acc10);
    acc11 = wmma_bf(a1, b1, acc11);
  }

  int nlo = lane & 15, mhi = (lane & 16) ? 8 : 0;
#pragma unroll
  for (int mi = 0; mi < 2; ++mi) {
#pragma unroll
    for (int ni = 0; ni < 2; ++ni) {
      v8f acc = (mi == 0) ? (ni == 0 ? acc00 : acc01)
                          : (ni == 0 ? acc10 : acc11);
      int gn = tile_n + ng + ni * 16 + nlo;
      int mrow = tile_m + mg + mi * 16 + mhi;
#pragma unroll
      for (int v = 0; v < 8; ++v) {
        int gm = mrow + v;
        if (gm >= M) continue;
        float x = acc[v];
        if (accum) x += outF[(size_t)gm * N + gn];
        if (bias) x += bias[gn];
        if (act == 1) x = fmaxf(x, 0.f);
        if (outF) outF[(size_t)gm * N + gn] = x;
        if (outB) outB[(size_t)gm * N + gn] = (bf16_t)x;
      }
    }
  }
}

// ---------------------------------------------------------------------------
// Fused GRU iteration: one workgroup = 16 messages, 512 threads = 16 waves.
// LDS holds the 96 gathered neighbor rows (bf16, L2-fed), sum_h, sum_gh.
// Each wave owns two 16-column sub-tiles: z-GEMM, the 6-neighbor r-GEMM with
// the U_r fragment loaded ONCE per K-step (6 accumulators), pre_h-GEMM and
// the GRU combine — all through v_wmma bf16.
// ---------------------------------------------------------------------------
__global__ __launch_bounds__(512) void gru_iter_kernel(
    const bf16_t* __restrict__ h_in, bf16_t* __restrict__ h_out,
    const int* __restrict__ bgraph,
    const float* __restrict__ rx, const float* __restrict__ zx,
    const float* __restrict__ hx,
    const bf16_t* __restrict__ WzT, const bf16_t* __restrict__ UrT,
    const bf16_t* __restrict__ WhT, const float* __restrict__ b_r, int M) {
  __shared__ __align__(16) bf16_t hn[96 * 520];   // row = mi*6 + j
  __shared__ __align__(16) bf16_t sh[16 * 520];   // sum_h
  __shared__ __align__(16) bf16_t sg[16 * 520];   // sum_gh
  __shared__ int bg[96];

  int tid = threadIdx.x;
  int m0 = blockIdx.x * 16;
  if (tid < 96) {
    int mi = tid / 6, j = tid % 6;
    int gm = m0 + mi;
    bg[tid] = (gm < M) ? bgraph[(size_t)gm * 6 + j] : 0;
  }
  __syncthreads();

  // Gather 96 rows x 512 bf16 from L2-resident h (16B chunks; 12 per thread).
  for (int i = tid; i < 96 * 64; i += 512) {
    int r = i >> 6, c8 = (i & 63) << 3;
    *(uint4*)&hn[r * 520 + c8] = *(const uint4*)(h_in + (size_t)bg[r] * HS + c8);
  }
  __syncthreads();

  // sum_h = sum over 6 neighbors (vectorized 4 cols / thread-step)
  for (int i = tid; i < 16 * 128; i += 512) {
    int mi = i >> 7, c4 = (i & 127) << 2;
    float s0 = 0.f, s1 = 0.f, s2 = 0.f, s3 = 0.f;
#pragma unroll
    for (int j = 0; j < 6; ++j) {
      v4bf x = *(const v4bf*)&hn[(mi * 6 + j) * 520 + c4];
      s0 += (float)x[0]; s1 += (float)x[1]; s2 += (float)x[2]; s3 += (float)x[3];
    }
    v4bf o;
    o[0] = (bf16_t)s0; o[1] = (bf16_t)s1; o[2] = (bf16_t)s2; o[3] = (bf16_t)s3;
    *(v4bf*)&sh[mi * 520 + c4] = o;
  }
  __syncthreads();

  int wave = tid >> 5, lane = tid & 31;
  int nlo = lane & 15, mhi = (lane & 16) ? 8 : 0;
  v8f zacc[2];

#pragma unroll
  for (int t = 0; t < 2; ++t) {
    int ns = wave * 32 + t * 16;
    int gn = ns + nlo;

    // z = sigmoid(zx + sum_h @ Wz_bot)
    v8f acc = zero8();
    for (int ks = 0; ks < 16; ++ks) {
      v16bf a = load_frag(&sh[ks * 32], 520, lane);
      v16bf b = load_frag(WzT + (size_t)ns * HS + ks * 32, HS, lane);
      acc = wmma_bf(a, b, acc);
    }
#pragma unroll
    for (int v = 0; v < 8; ++v) {
      int gm = m0 + mhi + v;
      float x = acc[v] + ((gm < M) ? zx[(size_t)gm * HS + gn] : 0.f);
      acc[v] = 1.f / (1.f + __expf(-x));
    }
    zacc[t] = acc;

    // r_j = sigmoid(rx + h_nei_j @ U_r + b_r), j=0..5, B-frag loaded once/K-step
    v8f racc[6];
#pragma unroll
    for (int j = 0; j < 6; ++j) racc[j] = zero8();
    for (int ks = 0; ks < 16; ++ks) {
      v16bf b = load_frag(UrT + (size_t)ns * HS + ks * 32, HS, lane);
#pragma unroll
      for (int j = 0; j < 6; ++j) {
        v16bf a = load_frag(&hn[j * 520 + ks * 32], 6 * 520, lane);
        racc[j] = wmma_bf(a, b, racc[j]);
      }
    }
    // gated neighbor sum: sum_gh = sum_j sigmoid(r_j) * h_nei_j
    float xr[8];
    float brn = b_r[gn];
#pragma unroll
    for (int v = 0; v < 8; ++v) {
      int gm = m0 + mhi + v;
      xr[v] = brn + ((gm < M) ? rx[(size_t)gm * HS + gn] : 0.f);
    }
    v8f g = zero8();
#pragma unroll
    for (int j = 0; j < 6; ++j) {
#pragma unroll
      for (int v = 0; v < 8; ++v) {
        float rs = 1.f / (1.f + __expf(-(racc[j][v] + xr[v])));
        g[v] += rs * (float)hn[((mhi + v) * 6 + j) * 520 + gn];
      }
    }
#pragma unroll
    for (int v = 0; v < 8; ++v) sg[(mhi + v) * 520 + gn] = (bf16_t)g[v];
  }
  __syncthreads();

  // pre_h = tanh(hx + sum_gh @ Wh_bot); h = ((1-z)*sum_h + z*pre_h)*mask
#pragma unroll
  for (int t = 0; t < 2; ++t) {
    int ns = wave * 32 + t * 16;
    int gn = ns + nlo;
    v8f acc = zero8();
    for (int ks = 0; ks < 16; ++ks) {
      v16bf a = load_frag(&sg[ks * 32], 520, lane);
      v16bf b = load_frag(WhT + (size_t)ns * HS + ks * 32, HS, lane);
      acc = wmma_bf(a, b, acc);
    }
#pragma unroll
    for (int v = 0; v < 8; ++v) {
      int gm = m0 + mhi + v;
      if (gm >= M) continue;
      float x = acc[v] + hx[(size_t)gm * HS + gn];
      float e = __expf(2.f * x);
      float ph = (e - 1.f) / (e + 1.f);
      float z = zacc[t][v];
      float hnew = (1.f - z) * (float)sh[(mhi + v) * 520 + gn] + z * ph;
      if (gm == 0) hnew = 0.f;  // pad message stays zero
      h_out[(size_t)gm * HS + gn] = (bf16_t)hnew;
    }
  }
}

// ---------------------------------------------------------------------------
// nei[a] = sum_j h[agraph[a][j]]  (bf16 gather-sum, L2-resident h)
// ---------------------------------------------------------------------------
__global__ __launch_bounds__(256) void nei_kernel(const bf16_t* __restrict__ h,
                                                  const int* __restrict__ agraph,
                                                  bf16_t* __restrict__ nei) {
  int a = blockIdx.x;
  int idx[6];
#pragma unroll
  for (int j = 0; j < 6; ++j) idx[j] = agraph[(size_t)a * 6 + j];
  for (int c = threadIdx.x; c < HS; c += 256) {
    float s = 0.f;
#pragma unroll
    for (int j = 0; j < 6; ++j) s += (float)h[(size_t)idx[j] * HS + c];
    nei[(size_t)a * HS + c] = (bf16_t)s;
  }
}

// ---------------------------------------------------------------------------
// Scoped attention core: one workgroup per (mol, head), 48x48 softmax in LDS.
// ---------------------------------------------------------------------------
__global__ __launch_bounds__(256) void attn_kernel(const float* __restrict__ q,
                                                   const float* __restrict__ k,
                                                   const float* __restrict__ v,
                                                   bf16_t* __restrict__ o) {
  __shared__ float qs[48][64], ks_[48][64], vs[48][64];
  __shared__ float ps[48][49];
  int mol = blockIdx.x, head = blockIdx.y;
  int tid = threadIdx.x;
  size_t base = (size_t)mol * LMOL * HS + head * DHEAD;
  for (int i = tid; i < 48 * 64; i += 256) {
    int r = i >> 6, c = i & 63;
    size_t off = base + (size_t)r * HS + c;
    qs[r][c] = q[off]; ks_[r][c] = k[off]; vs[r][c] = v[off];
  }
  __syncthreads();
  for (int i = tid; i < 48 * 48; i += 256) {
    int r = i / 48, c = i % 48;
    float s = 0.f;
#pragma unroll 8
    for (int d = 0; d < 64; ++d) s += qs[r][d] * ks_[c][d];
    ps[r][c] = s * 0.125f;  // d^-0.5, d=64
  }
  __syncthreads();
  if (tid < 48) {
    float mx = -1e30f;
    for (int j = 0; j < 48; ++j) mx = fmaxf(mx, ps[tid][j]);
    float ssum = 0.f;
    for (int j = 0; j < 48; ++j) { float e = __expf(ps[tid][j] - mx); ps[tid][j] = e; ssum += e; }
    float inv = 1.f / ssum;
    for (int j = 0; j < 48; ++j) ps[tid][j] *= inv;
  }
  __syncthreads();
  for (int i = tid; i < 48 * 64; i += 256) {
    int r = i >> 6, c = i & 63;
    float s = 0.f;
#pragma unroll 8
    for (int j = 0; j < 48; ++j) s += ps[r][j] * vs[j][c];
    o[base + (size_t)r * HS + c] = (bf16_t)s;
  }
}

// ---------------------------------------------------------------------------
// LayerNorm (+optional residual) over 512-wide rows.
// ---------------------------------------------------------------------------
__global__ __launch_bounds__(256) void ln_kernel(
    const float* __restrict__ x, const float* __restrict__ resid,
    const float* __restrict__ g, const float* __restrict__ b,
    float* __restrict__ outF, bf16_t* __restrict__ outB) {
  __shared__ float red[256];
  int row = blockIdx.x, tid = threadIdx.x;
  const float* xr = x + (size_t)row * HS;
  float s = 0.f;
  for (int c = tid; c < HS; c += 256) s += xr[c];
  red[tid] = s; __syncthreads();
  for (int st = 128; st > 0; st >>= 1) { if (tid < st) red[tid] += red[tid + st]; __syncthreads(); }
  float mean = red[0] / (float)HS;
  __syncthreads();
  float vv = 0.f;
  for (int c = tid; c < HS; c += 256) { float d = xr[c] - mean; vv += d * d; }
  red[tid] = vv; __syncthreads();
  for (int st = 128; st > 0; st >>= 1) { if (tid < st) red[tid] += red[tid + st]; __syncthreads(); }
  float inv = rsqrtf(red[0] / (float)HS + 1e-5f);
  for (int c = tid; c < HS; c += 256) {
    float o = (xr[c] - mean) * inv * g[c] + b[c];
    if (resid) o += resid[(size_t)row * HS + c];
    if (outF) outF[(size_t)row * HS + c] = o;
    if (outB) outB[(size_t)row * HS + c] = (bf16_t)o;
  }
}

// hmol[mol] = sum over 48 atoms of hatom_out (segment-sum, fixed scopes)
__global__ __launch_bounds__(256) void segsum_kernel(const float* __restrict__ hatom,
                                                     float* __restrict__ hmol) {
  int mol = blockIdx.x;
  int c = blockIdx.y * 256 + threadIdx.x;
  float s = 0.f;
  for (int i = 0; i < LMOL; ++i)
    s += hatom[(size_t)(1 + mol * LMOL + i) * HS + c];
  hmol[(size_t)mol * HS + c] = s;
}

// ---------------------------------------------------------------------------
// Host orchestration
// ---------------------------------------------------------------------------
extern "C" void kernel_launch(void* const* d_in, const int* in_sizes, int n_in,
                              void* d_out, int out_size, void* d_ws, size_t ws_size,
                              hipStream_t stream) {
  (void)in_sizes; (void)out_size; (void)ws_size;
  if (n_in < 41) return;

  const float* fnode  = (const float*)d_in[0];   // [NA, 98]
  const float* fmess  = (const float*)d_in[1];   // [MMSG, 104]
  const int*   agraph = (const int*)d_in[2];     // [NA, 6]
  const int*   bgraph = (const int*)d_in[3];     // [MMSG, 6]
  // d_in[4] = mol_ids (implicit in fixed 48-atom scopes)

  // params flattened in sorted-key pytree order:
  // blocks[b]: W_o, attn{Wk,Wo,Wq,Wv}, b_o, gru{U_r,W_h,W_r,W_z,b_h,b_r,b_z}, ln_b, ln_g
  const float *PW_o[2], *PWk[2], *PWoA[2], *PWq[2], *PWv[2], *Pb_o[2];
  const float *PUr[2], *PWh[2], *PWr[2], *PWz[2], *Pb_h[2], *Pb_r[2], *Pb_z[2];
  const float *Pln_b[2], *Pln_g[2];
  for (int b = 0; b < 2; ++b) {
    int s = 5 + b * 15;
    PW_o[b]  = (const float*)d_in[s + 0];
    PWk[b]   = (const float*)d_in[s + 1];
    PWoA[b]  = (const float*)d_in[s + 2];
    PWq[b]   = (const float*)d_in[s + 3];
    PWv[b]   = (const float*)d_in[s + 4];
    Pb_o[b]  = (const float*)d_in[s + 5];
    PUr[b]   = (const float*)d_in[s + 6];
    PWh[b]   = (const float*)d_in[s + 7];
    PWr[b]   = (const float*)d_in[s + 8];
    PWz[b]   = (const float*)d_in[s + 9];
    Pb_h[b]  = (const float*)d_in[s + 10];
    Pb_r[b]  = (const float*)d_in[s + 11];
    Pb_z[b]  = (const float*)d_in[s + 12];
    Pln_b[b] = (const float*)d_in[s + 13];
    Pln_g[b] = (const float*)d_in[s + 14];
  }
  const float* W1      = (const float*)d_in[35];  // [610, 1024]
  const float* W2      = (const float*)d_in[36];  // [1024, 512]
  const float* b1      = (const float*)d_in[37];
  const float* b2      = (const float*)d_in[38];
  const float* lnout_b = (const float*)d_in[39];
  const float* lnout_g = (const float*)d_in[40];

  // ---- workspace bump allocator (same order every call: deterministic) ----
  char* ws = (char*)d_ws;
  size_t off = 0;
  auto alloc = [&](size_t bytes) -> void* {
    void* p = ws + off;
    off = (off + bytes + 255) & ~(size_t)255;
    return p;
  };

  auto wconv = [&](const float* in, int row_off, int ldin, int K, int N,
                   int Kp) -> bf16_t* {
    bf16_t* o = (bf16_t*)alloc((size_t)N * Kp * sizeof(bf16_t));
    int total = N * Kp;
    wconv_kernel<<<(total + 255) / 256, 256, 0, stream>>>(
        in + (size_t)row_off * ldin, ldin, K, N, o, Kp);
    return o;
  };
  auto gemm = [&](const bf16_t* A, int lda, int M, int K, const bf16_t* BT,
                  int N, const float* bias, int act, int accum,
                  float* outF, bf16_t* outB) {
    dim3 g((M + 63) / 64, N / 128);
    gemm_bf16_kernel<<<g, 256, 0, stream>>>(A, lda, M, K, BT, K, N, bias, act,
                                            accum, outF, outB);
  };
  auto zero = [&](void* p, size_t bytes) {
    long n = (long)(bytes / 4);
    zero32_kernel<<<(unsigned)((n + 255) / 256), 256, 0, stream>>>(
        (unsigned int*)p, n);
  };

  // ---- weights -> bf16, transposed [N][Kpad] ----
  bf16_t *WrT[2], *WzTopT[2], *WzBotT[2], *WhTopT[2], *WhBotT[2], *UrT[2];
  bf16_t *WoTopT[2], *WoBotT[2], *WqT[2], *WkT[2], *WvT[2], *WoAT[2];
  for (int b = 0; b < 2; ++b) {
    int nf = (b == 0) ? 98 : 512;
    int nfp = (b == 0) ? 128 : 512;
    WrT[b]    = wconv(PWr[b], 0, HS, 104, HS, 128);
    WzTopT[b] = wconv(PWz[b], 0, HS, 104, HS, 128);
    WzBotT[b] = wconv(PWz[b], 104, HS, HS, HS, HS);
    WhTopT[b] = wconv(PWh[b], 0, HS, 104, HS, 128);
    WhBotT[b] = wconv(PWh[b], 104, HS, HS, HS, HS);
    UrT[b]    = wconv(PUr[b], 0, HS, HS, HS, HS);
    WoTopT[b] = wconv(PW_o[b], 0, HS, nf, HS, nfp);
    WoBotT[b] = wconv(PW_o[b], nf, HS, HS, HS, HS);
    WqT[b]    = wconv(PWq[b], 0, HS, HS, HS, HS);
    WkT[b]    = wconv(PWk[b], 0, HS, HS, HS, HS);
    WvT[b]    = wconv(PWv[b], 0, HS, HS, HS, HS);
    WoAT[b]   = wconv(PWoA[b], 0, HS, HS, HS, HS);
  }
  bf16_t* W1TopT = wconv(W1, 0, 1024, 512, 1024, 512);
  bf16_t* W1BotT = wconv(W1, 512, 1024, 98, 1024, 128);
  bf16_t* W2T    = wconv(W2, 0, HS, 1024, HS, 1024);

  // ---- activation buffers (A operands padded: rows to 64, K to 32) ----
  bf16_t* fmess_bf  = (bf16_t*)alloc((size_t)MSG_PAD * 128 * 2);
  bf16_t* fnode0_bf = (bf16_t*)alloc((size_t)NA_PAD * 128 * 2);
  bf16_t* h_a       = (bf16_t*)alloc((size_t)MMSG * HS * 2);
  bf16_t* h_b       = (bf16_t*)alloc((size_t)MMSG * HS * 2);
  float*  rx        = (float*)alloc((size_t)MMSG * HS * 4);  // also: q, ffn tmp f32
  float*  zx        = (float*)alloc((size_t)MMSG * HS * 4);  // also: k
  float*  hx        = (float*)alloc((size_t)MMSG * HS * 4);  // also: v, ffn out f32
  bf16_t* nei_bf    = (bf16_t*)alloc((size_t)NA_PAD * HS * 2);
  float*  hatom     = (float*)alloc((size_t)NA * HS * 4);
  bf16_t* hatom_bf  = (bf16_t*)alloc((size_t)NA_PAD * HS * 2);
  bf16_t* xnode_bf  = (bf16_t*)alloc((size_t)NA_PAD * HS * 2);
  bf16_t* o_bf      = (bf16_t*)alloc((size_t)NATT * HS * 2);  // 24576 % 64 == 0
  float*  attnf     = (float*)alloc((size_t)NA * HS * 4);
  bf16_t* tmp_bf    = (bf16_t*)alloc((size_t)NA_PAD * 1024 * 2);

  convpad_kernel<<<(unsigned)(((long)MSG_PAD * 128 + 255) / 256), 256, 0, stream>>>(
      fmess, fmess_bf, MMSG, 104, 128, (long)MSG_PAD * 128);
  convpad_kernel<<<(unsigned)(((long)NA_PAD * 128 + 255) / 256), 256, 0, stream>>>(
      fnode, fnode0_bf, NA, 98, 128, (long)NA_PAD * 128);

  // zero the 63 pad rows of writer-produced A operands (once per call)
  zero(nei_bf + (size_t)NA * HS, (size_t)(NA_PAD - NA) * HS * 2);
  zero(hatom_bf + (size_t)NA * HS, (size_t)(NA_PAD - NA) * HS * 2);
  zero(xnode_bf + (size_t)NA * HS, (size_t)(NA_PAD - NA) * HS * 2);
  zero(tmp_bf + (size_t)NA * 1024, (size_t)(NA_PAD - NA) * 1024 * 2);

  float* dout = (float*)d_out;
  float* hmol_out  = dout;                       // [512, 512]
  float* hatom_out = dout + (size_t)NMOL * HS;   // [NA, 512]

  const int gru_tiles = (MMSG + 15) / 16;

  for (int b = 0; b < 2; ++b) {
    const bf16_t* feat_bf = (b == 0) ? fnode0_bf : xnode_bf;
    int ldf = (b == 0) ? 128 : 512;  // K padded into the buffers

    // loop-invariant message projections (K=128 zero-padded)
    gemm(fmess_bf, 128, MMSG, 128, WrT[b], HS, nullptr, 0, 0, rx, nullptr);
    gemm(fmess_bf, 128, MMSG, 128, WzTopT[b], HS, Pb_z[b], 0, 0, zx, nullptr);
    gemm(fmess_bf, 128, MMSG, 128, WhTopT[b], HS, Pb_h[b], 0, 0, hx, nullptr);

    // GRU message passing, DEPTH=8, double-buffered h in bf16 (L2-resident)
    zero(h_a, (size_t)MMSG * HS * 2);
    bf16_t* hin = h_a; bf16_t* hout = h_b;
    for (int d = 0; d < 8; ++d) {
      gru_iter_kernel<<<gru_tiles, 512, 0, stream>>>(
          hin, hout, bgraph, rx, zx, hx, WzBotT[b], UrT[b], WhBotT[b],
          Pb_r[b], MMSG);
      bf16_t* t = hin; hin = hout; hout = t;
    }
    bf16_t* h_fin = hin;  // after even #iters

    // hatom = relu([feat, nei] @ W_o + b_o)
    nei_kernel<<<NA, 256, 0, stream>>>(h_fin, agraph, nei_bf);
    gemm(nei_bf, HS, NA, HS, WoBotT[b], HS, nullptr, 0, 0, hatom, nullptr);
    gemm(feat_bf, ldf, NA, ldf, WoTopT[b], HS, Pb_o[b], 1, 1, hatom, hatom_bf);

    // scoped MHA on hatom[1:]
    gemm(hatom_bf + HS, HS, NATT, HS, WqT[b], HS, nullptr, 0, 0, rx, nullptr);
    gemm(hatom_bf + HS, HS, NATT, HS, WkT[b], HS, nullptr, 0, 0, zx, nullptr);
    gemm(hatom_bf + HS, HS, NATT, HS, WvT[b], HS, nullptr, 0, 0, hx, nullptr);
    attn_kernel<<<dim3(NMOL, NHEAD), 256, 0, stream>>>(rx, zx, hx, o_bf);
    zero(attnf, HS * 4);  // attn_full row 0
    gemm(o_bf, HS, NATT, HS, WoAT[b], HS, nullptr, 0, 0, attnf + HS, nullptr);

    // x_next = hatom + LN(attn_full)*g + b
    ln_kernel<<<NA, 256, 0, stream>>>(attnf, hatom, Pln_g[b], Pln_b[b], nullptr,
                                      xnode_bf);
  }

  // FFN on [h, fnode_orig], then LN, then segment-sum
  gemm(xnode_bf, HS, NA, HS, W1TopT, 1024, nullptr, 0, 0, rx, nullptr);
  gemm(fnode0_bf, 128, NA, 128, W1BotT, 1024, b1, 1, 1, rx, tmp_bf);
  gemm(tmp_bf, 1024, NA, 1024, W2T, HS, b2, 0, 0, hx, nullptr);
  ln_kernel<<<NA, 256, 0, stream>>>(hx, nullptr, lnout_g, lnout_b, hatom_out,
                                    nullptr);
  segsum_kernel<<<dim3(NMOL, 2), 256, 0, stream>>>(hatom_out, hmol_out);
}
